// ManifoldEncoderLayer_90220083020132
// MI455X (gfx1250) — compile-verified, matrix operand order checked by reading, not run on hardware
//
#include <hip/hip_runtime.h>
#include <hip/hip_bf16.h>
#include <math.h>
#include <stdint.h>

typedef __attribute__((ext_vector_type(16))) _Float16 v16h;
typedef __attribute__((ext_vector_type(8)))  _Float16 v8h;
typedef __attribute__((ext_vector_type(8)))  float    v8f;
typedef __attribute__((ext_vector_type(4)))  unsigned int v4u;
typedef __attribute__((ext_vector_type(8)))  int v8i;
typedef __attribute__((ext_vector_type(4)))  int v4i;

// Problem dims: B=16, N=512, C=384, H=6, HD=64, FF=1152; tokens=8192; BH=96.

// ---------------------------------------------------------------------------
// Tensor Data Mover: 2-D tile load Global -> LDS.
// D# per CDNA5 ISA ch.8: group0 = {count=1 | lds_addr | global_addr(57b) | type=2},
// group1 = {data_size=2B, tensor_dim0=tile_k, tensor_dim1=rows, tile_dim0=tile_k,
//           tile_dim1=rows, tensor_dim0_stride=ld}.  Tile start == global_addr, so
// the tile is fully in-bounds by construction.  Tracked by TENSORcnt.
// ---------------------------------------------------------------------------
__device__ inline void tdm_load_2d(unsigned lds_byte_addr, const void* gptr,
                                   unsigned rows, unsigned tile_k, unsigned stride_elems)
{
    unsigned long long ga = (unsigned long long)(uintptr_t)gptr;
    v4u g0;
    g0[0] = 1u;                                          // count=1, user load
    g0[1] = lds_byte_addr;                               // LDS destination (bytes)
    g0[2] = (unsigned)(ga & 0xFFFFFFFFu);                // global addr [31:0]
    g0[3] = (unsigned)((ga >> 32) & 0x01FFFFFFu)         // global addr [56:32]
          | (2u << 30);                                  // type = 2 ("image")
    v8i g1;
    g1[0] = (int)(1u << 16);                             // data_size = 1 -> 2 bytes
    g1[1] = (int)((tile_k & 0xFFFFu) << 16);             // tensor_dim0[15:0] @ bits 63:48
    g1[2] = (int)(((tile_k >> 16) & 0xFFFFu)             // tensor_dim0[31:16]
          | ((rows & 0xFFFFu) << 16));                   // tensor_dim1[15:0]
    g1[3] = (int)(((rows >> 16) & 0xFFFFu)               // tensor_dim1[31:16]
          | ((tile_k & 0xFFFFu) << 16));                 // tile_dim0
    g1[4] = (int)(rows & 0xFFFFu);                       // tile_dim1 (tile_dim2 = 0)
    g1[5] = (int)stride_elems;                           // tensor_dim0_stride[31:0]
    g1[6] = 0;                                           // stride hi / dim1_stride lo
    g1[7] = 0;
    v4i z4 = {0, 0, 0, 0};
#if __clang_major__ >= 23
    v8i z8 = {0, 0, 0, 0, 0, 0, 0, 0};
    __builtin_amdgcn_tensor_load_to_lds(g0, g1, z4, z4, z8, 0);
#else
    __builtin_amdgcn_tensor_load_to_lds(g0, g1, z4, z4, 0);
#endif
}

// Generic shared pointer -> LDS byte offset (ISA 10.2: LDS aperture addr = addr[31:0]).
__device__ inline unsigned lds_off(const void* p) { return (unsigned)(uintptr_t)p; }

// ---------------------------------------------------------------------------
// Block-tiled batched WMMA GEMM with TDM staging:
//   C[z][m][n] = sum_k A[z][m][k] * W[z][n][k]   (A, W row-major f16)
// Block = 256 threads (8 waves) computes a 64m x 128n macro tile; wave (mw,nw)
// holds a 32x32 register tile = 4 f32 accumulators, 4 WMMAs per 32-wide K-step.
// Wave 0 DMAs the A (64x32) and B (128x32) tiles into LDS via TENSOR_LOAD_TO_LDS,
// waits on TENSORcnt, and the workgroup barrier publishes them.
// Requires: M % 64 == 0, N % 128 == 0, K % 32 == 0.
// ---------------------------------------------------------------------------
__global__ void __launch_bounds__(256)
gemm_wmma_tdm(const _Float16* __restrict__ A, long long sAb, int lda,
              const _Float16* __restrict__ W, long long sWb, int ldw,
              float* __restrict__ outF, _Float16* __restrict__ outH,
              int outZDiv, long long sOouter, long long sOinner, int ldo,
              const float* __restrict__ bias, const float* __restrict__ resid,
              int K, int act)
{
    __shared__ __align__(16) _Float16 As[64][32];
    __shared__ __align__(16) _Float16 Bs[128][32];

    const int lane = threadIdx.x & 31;
    const int wv   = threadIdx.x >> 5;
    const int mw   = wv >> 2;            // 0..1  (32-row half of the macro tile)
    const int nw   = wv & 3;             // 0..3  (32-col quarter)
    const long long z = blockIdx.z;
    const int m0 = blockIdx.y * 64;
    const int n0 = blockIdx.x * 128;

    const _Float16* Ab = A + z * sAb + (long long)m0 * lda;
    const _Float16* Wb = W + z * sWb + (long long)n0 * ldw;
    const unsigned ldsA = lds_off(&As[0][0]);
    const unsigned ldsB = lds_off(&Bs[0][0]);

    const int r16  = lane & 15;
    const int kgrp = (lane >> 4) << 3;   // lanes 0-15: K 0..7/16..23 ; 16-31: 8..15/24..31

    v8f acc[2][2] = {};
    for (int k0 = 0; k0 < K; k0 += 32) {
        if (wv == 0) {
            tdm_load_2d(ldsA, (const void*)(Ab + k0), 64,  32, (unsigned)lda);
            tdm_load_2d(ldsB, (const void*)(Wb + k0), 128, 32, (unsigned)ldw);
            __builtin_amdgcn_s_wait_tensorcnt(0);
        }
        __syncthreads();

        v16h af[2], bf[2];
        #pragma unroll
        for (int s = 0; s < 2; ++s) {
            v8h lo = *(const v8h*)(&As[mw * 32 + s * 16 + r16][kgrp]);
            v8h hi = *(const v8h*)(&As[mw * 32 + s * 16 + r16][kgrp + 16]);
            #pragma unroll
            for (int i = 0; i < 8; ++i) { af[s][i] = lo[i]; af[s][8 + i] = hi[i]; }
        }
        #pragma unroll
        for (int s = 0; s < 2; ++s) {
            v8h lo = *(const v8h*)(&Bs[nw * 32 + s * 16 + r16][kgrp]);
            v8h hi = *(const v8h*)(&Bs[nw * 32 + s * 16 + r16][kgrp + 16]);
            #pragma unroll
            for (int i = 0; i < 8; ++i) { bf[s][i] = lo[i]; bf[s][8 + i] = hi[i]; }
        }
        #pragma unroll
        for (int i = 0; i < 2; ++i)
            #pragma unroll
            for (int j = 0; j < 2; ++j)
                acc[i][j] = __builtin_amdgcn_wmma_f32_16x16x32_f16(
                    false, af[i], false, bf[j], (short)0, acc[i][j], false, false);
        __syncthreads();
    }

    const long long zo = (z / outZDiv) * sOouter + (z % outZDiv) * sOinner;
    #pragma unroll
    for (int i = 0; i < 2; ++i) {
        const int rbase = m0 + mw * 32 + i * 16 + ((lane >> 4) << 3);
        #pragma unroll
        for (int j = 0; j < 2; ++j) {
            const int col = n0 + nw * 32 + j * 16 + r16;
            const float bval = bias ? bias[col] : 0.0f;
            #pragma unroll
            for (int r = 0; r < 8; ++r) {
                float v = acc[i][j][r] + bval;
                if (act == 1) v = 0.5f * v * (1.0f + erff(v * 0.70710678118654752f));
                long long idx = zo + (long long)(rbase + r) * ldo + col;
                if (resid) v += resid[idx];
                if (outF) outF[idx] = v;
                if (outH) outH[idx] = (_Float16)v;
            }
        }
    }
}

// ---------------------------------------------------------------------------
// Fallback wave-per-tile WMMA GEMM (used for the small-N batched GEMMs:
// qgr/kgr with N=64 and attn@V with N=64).  Same math contract as above.
// ---------------------------------------------------------------------------
__global__ void gemm_wmma_f16(
    const _Float16* __restrict__ A, long long sAb, int lda,
    const _Float16* __restrict__ W, long long sWb, int ldw,
    float* __restrict__ outF, _Float16* __restrict__ outH,
    int outZDiv, long long sOouter, long long sOinner, int ldo,
    const float* __restrict__ bias, const float* __restrict__ resid,
    int M, int N, int K, int act)
{
    const int lane = threadIdx.x & 31;
    const int wv   = threadIdx.x >> 5;
    const int tn   = blockIdx.x * 4 + wv;
    const int tm   = blockIdx.y;
    if (tn * 16 >= N) return;
    const long long z = blockIdx.z;

    const _Float16* Ab = A + z * sAb;
    const _Float16* Wb = W + z * sWb;

    const int r16  = lane & 15;
    const int kgrp = (lane >> 4) << 3;
    const _Float16* arow = Ab + (long long)(tm * 16 + r16) * lda + kgrp;
    const _Float16* wrow = Wb + (long long)(tn * 16 + r16) * ldw + kgrp;

    v8f acc = {};
    #pragma unroll 2
    for (int k0 = 0; k0 < K; k0 += 32) {
        if (k0 + 64 < K) {
            __builtin_prefetch(arow + k0 + 64, 0, 1);
            __builtin_prefetch(wrow + k0 + 64, 0, 1);
        }
        v8h a0 = *(const v8h*)(arow + k0);
        v8h a1 = *(const v8h*)(arow + k0 + 16);
        v8h b0 = *(const v8h*)(wrow + k0);
        v8h b1 = *(const v8h*)(wrow + k0 + 16);
        v16h av, bv;
        #pragma unroll
        for (int i = 0; i < 8; ++i) {
            av[i] = a0[i]; av[8 + i] = a1[i];
            bv[i] = b0[i]; bv[8 + i] = b1[i];
        }
        acc = __builtin_amdgcn_wmma_f32_16x16x32_f16(
            false, av, false, bv, (short)0, acc, false, false);
    }

    const long long zo = (z / outZDiv) * sOouter + (z % outZDiv) * sOinner;
    const int col   = tn * 16 + r16;
    const int rbase = tm * 16 + ((lane >> 4) << 3);
    const float bval = bias ? bias[col] : 0.0f;
    #pragma unroll
    for (int r = 0; r < 8; ++r) {
        float v = acc[r] + bval;
        if (act == 1) v = 0.5f * v * (1.0f + erff(v * 0.70710678118654752f));
        long long idx = zo + (long long)(rbase + r) * ldo + col;
        if (resid) v += resid[idx];
        if (outF) outF[idx] = v;
        if (outH) outH[idx] = (_Float16)v;
    }
}

// ---------------------------------------------------------------------------
// Fused (optional residual add) + LayerNorm over C=384, writes f16 (+opt f32)
// ---------------------------------------------------------------------------
__global__ void add_ln_kernel(const float* __restrict__ x, const float* __restrict__ add,
                              const float* __restrict__ w, const float* __restrict__ bw,
                              _Float16* __restrict__ outH, float* __restrict__ outF)
{
    const int token = blockIdx.x;
    const int t = threadIdx.x;
    const long long base = (long long)token * 384;
    float xv[3];
    #pragma unroll
    for (int i = 0; i < 3; ++i) {
        int c = t + 128 * i;
        xv[i] = x[base + c] + (add ? add[base + c] : 0.0f);
    }
    __shared__ float red[128];
    red[t] = xv[0] + xv[1] + xv[2];
    __syncthreads();
    for (int st = 64; st > 0; st >>= 1) { if (t < st) red[t] += red[t + st]; __syncthreads(); }
    const float mean = red[0] * (1.0f / 384.0f);
    __syncthreads();
    float vs = 0.0f;
    #pragma unroll
    for (int i = 0; i < 3; ++i) { float d = xv[i] - mean; vs += d * d; }
    red[t] = vs;
    __syncthreads();
    for (int st = 64; st > 0; st >>= 1) { if (t < st) red[t] += red[t + st]; __syncthreads(); }
    const float rstd = rsqrtf(red[0] * (1.0f / 384.0f) + 1e-5f);
    #pragma unroll
    for (int i = 0; i < 3; ++i) {
        int c = t + 128 * i;
        float o = (xv[i] - mean) * rstd * w[c] + bw[c];
        outH[base + c] = (_Float16)o;
        if (outF) outF[base + c] = o;
    }
}

// ---------------------------------------------------------------------------
// Split fused qkv output [token,1152] into qh/kh [b,h,n,d], vTh [b,h,d,n],
// and per-token head norms |q|^2, |k|^2.  grid = 8192, block = 384
// ---------------------------------------------------------------------------
__global__ void qkv_rearrange_kernel(const _Float16* __restrict__ qkv,
                                     _Float16* __restrict__ qh, _Float16* __restrict__ kh,
                                     _Float16* __restrict__ vTh,
                                     float* __restrict__ qn2, float* __restrict__ kn2)
{
    const int token = blockIdx.x;
    const int b = token >> 9, n = token & 511;
    const int t = threadIdx.x;
    const int h = t >> 6, d = t & 63;
    const _Float16* row = qkv + (long long)token * 1152;
    _Float16 qv = row[t], kv = row[384 + t], vv = row[768 + t];
    long long bh = (long long)(b * 6 + h);
    qh[(bh * 512 + n) * 64 + d] = qv;
    kh[(bh * 512 + n) * 64 + d] = kv;
    vTh[(bh * 64 + d) * 512 + n] = vv;

    __shared__ float red[384];
    float qf = (float)qv, kf = (float)kv;
    red[t] = qf * qf; __syncthreads();
    for (int st = 32; st > 0; st >>= 1) { if (d < st) red[t] += red[t + st]; __syncthreads(); }
    if (d == 0) qn2[bh * 512 + n] = red[h * 64];
    __syncthreads();
    red[t] = kf * kf; __syncthreads();
    for (int st = 32; st > 0; st >>= 1) { if (d < st) red[t] += red[t + st]; __syncthreads(); }
    if (d == 0) kn2[bh * 512 + n] = red[h * 64];
}

// ---------------------------------------------------------------------------
// Gram matrix G = X^T X per (b,h): X = [512,64] f16 -> G [64,64] f32
// ---------------------------------------------------------------------------
__global__ void gram64_kernel(const _Float16* __restrict__ X, float* __restrict__ G)
{
    const long long bh = blockIdx.x;
    const _Float16* Xb = X + bh * 512 * 64;
    __shared__ _Float16 tile[64][65];
    float acc[16];
    #pragma unroll
    for (int i = 0; i < 16; ++i) acc[i] = 0.0f;
    for (int n0 = 0; n0 < 512; n0 += 64) {
        #pragma unroll
        for (int i = 0; i < 16; ++i) {
            int id = threadIdx.x + 256 * i;
            tile[id >> 6][id & 63] = Xb[(long long)(n0 + (id >> 6)) * 64 + (id & 63)];
        }
        __syncthreads();
        for (int i = 0; i < 16; ++i) {
            int id = threadIdx.x + 256 * i;
            int d = id >> 6, e = id & 63;
            float s = 0.0f;
            #pragma unroll 8
            for (int nn = 0; nn < 64; ++nn)
                s += (float)tile[nn][d] * (float)tile[nn][e];
            acc[i] += s;
        }
        __syncthreads();
    }
    #pragma unroll
    for (int i = 0; i < 16; ++i)
        G[bh * 4096 + threadIdx.x + 256 * i] = acc[i];
}

// ---------------------------------------------------------------------------
// Cholesky G = L L^T (64x64), then Linv = L^{-1}.  Cholesky-QR: Q = A R^{-1},
// R = L^T  =>  (R^{-1})^T = L^{-1}, stored row-major f16 as W operand of qgr GEMM.
// ---------------------------------------------------------------------------
__global__ void chol_inv64_kernel(const float* __restrict__ G, _Float16* __restrict__ RinvT)
{
    __shared__ float L[64][65];
    __shared__ float Inv[64][65];
    const int t = threadIdx.x;
    const float* Gb = G + (long long)blockIdx.x * 4096;
    for (int r = 0; r < 64; ++r) L[r][t] = Gb[r * 64 + t];
    __syncthreads();
    for (int k = 0; k < 64; ++k) {
        if (t == k) L[k][k] = sqrtf(fmaxf(L[k][k], 1e-20f));
        __syncthreads();
        if (t > k) L[t][k] = L[t][k] / L[k][k];
        __syncthreads();
        if (t > k) {
            float lik = L[t][k];
            for (int j = k + 1; j <= t; ++j) L[t][j] -= lik * L[j][k];
        }
        __syncthreads();
    }
    for (int i = 0; i < 64; ++i) {
        float s = (i == t) ? 1.0f : 0.0f;
        for (int j = t; j < i; ++j) s -= L[i][j] * Inv[j][t];
        Inv[i][t] = (i >= t) ? (s / L[i][i]) : 0.0f;
    }
    __syncthreads();
    for (int r = 0; r < 64; ++r)
        RinvT[(long long)blockIdx.x * 4096 + r * 64 + t] = (_Float16)Inv[r][t];
}

// ---------------------------------------------------------------------------
// Fused feature-build + conv-mix + softmax (one (b,n) query row per block).
// Writes f32 attn to d_out and f16 attn in-place over the grassmann buffer.
// ---------------------------------------------------------------------------
__global__ void mix_softmax_kernel(const _Float16* __restrict__ qkh,
                                   _Float16* __restrict__ grh,
                                   const float* __restrict__ qn2,
                                   const float* __restrict__ kn2,
                                   const float* __restrict__ conv_w,
                                   const float* __restrict__ conv_b,
                                   const float* __restrict__ scale_p,
                                   const float* __restrict__ riem_p,
                                   const float* __restrict__ grass_p,
                                   float* __restrict__ attn_out)
{
    const int n = blockIdx.x, b = blockIdx.y, t = threadIdx.x;
    __shared__ float smix[6][512];
    __shared__ float red[256];
    __shared__ float cw[108];
    __shared__ float cb[6];
    if (t < 108) cw[t] = conv_w[t];
    if (t < 6)   cb[t] = conv_b[t];
    __syncthreads();
    const float sc = scale_p[0], rs = riem_p[0], gs = grass_p[0];
    float qn4[6];
    #pragma unroll
    for (int h = 0; h < 6; ++h) {
        float q2 = qn2[((long long)(b * 6 + h)) * 512 + n];
        qn4[h] = q2 * q2;
    }
    #pragma unroll
    for (int half = 0; half < 2; ++half) {
        const int m = t + 256 * half;
        float mix[6];
        #pragma unroll
        for (int o = 0; o < 6; ++o) mix[o] = cb[o];
        #pragma unroll
        for (int h = 0; h < 6; ++h) {
            long long idx = (((long long)(b * 6 + h)) * 512 + n) * 512 + m;
            float qk = (float)qkh[idx];
            float gr = (float)grh[idx];
            float k2 = kn2[((long long)(b * 6 + h)) * 512 + m];
            float eucl  = qk * sc;
            float d2    = qn4[h] + k2 * k2 - 2.0f * qk * qk;
            float riem  = sqrtf(fmaxf(d2, 0.0f)) * rs;
            float grass = gr * gr * gs;
            #pragma unroll
            for (int o = 0; o < 6; ++o)
                mix[o] += cw[o * 18 + h] * eucl + cw[o * 18 + 6 + h] * riem
                        + cw[o * 18 + 12 + h] * grass;
        }
        #pragma unroll
        for (int o = 0; o < 6; ++o) smix[o][m] = mix[o];
    }
    __syncthreads();
    for (int o = 0; o < 6; ++o) {
        red[t] = fmaxf(smix[o][t], smix[o][t + 256]);
        __syncthreads();
        for (int st = 128; st > 0; st >>= 1) { if (t < st) red[t] = fmaxf(red[t], red[t + st]); __syncthreads(); }
        const float mx = red[0];
        __syncthreads();
        float e0 = __expf(smix[o][t] - mx);
        float e1 = __expf(smix[o][t + 256] - mx);
        red[t] = e0 + e1;
        __syncthreads();
        for (int st = 128; st > 0; st >>= 1) { if (t < st) red[t] += red[t + st]; __syncthreads(); }
        const float inv = 1.0f / red[0];
        __syncthreads();
        long long ob = (((long long)(b * 6 + o)) * 512 + n) * 512;
        attn_out[ob + t]       = e0 * inv;
        attn_out[ob + t + 256] = e1 * inv;
        grh[ob + t]            = (_Float16)(e0 * inv);
        grh[ob + t + 256]      = (_Float16)(e1 * inv);
    }
}

__global__ void cvt_f16_kernel(const float* __restrict__ in, _Float16* __restrict__ out, int n)
{
    int i = blockIdx.x * blockDim.x + threadIdx.x;
    if (i < n) out[i] = (_Float16)in[i];
}

// ---------------------------------------------------------------------------
static void launch_gemm_tdm(hipStream_t s,
                            const _Float16* A, long long sAb, int lda,
                            const _Float16* W, long long sWb, int ldw,
                            float* outF, _Float16* outH,
                            int zdiv, long long sOo, long long sOi, int ldo,
                            const float* bias, const float* resid,
                            int M, int N, int K, int act, int batch)
{
    dim3 grid(N / 128, M / 64, batch);
    gemm_wmma_tdm<<<grid, 256, 0, s>>>(A, sAb, lda, W, sWb, ldw, outF, outH,
                                       zdiv, sOo, sOi, ldo, bias, resid, K, act);
}

static void launch_gemm(hipStream_t s,
                        const _Float16* A, long long sAb, int lda,
                        const _Float16* W, long long sWb, int ldw,
                        float* outF, _Float16* outH,
                        int zdiv, long long sOo, long long sOi, int ldo,
                        const float* bias, const float* resid,
                        int M, int N, int K, int act, int batch)
{
    dim3 grid((N / 16 + 3) / 4, M / 16, batch);
    gemm_wmma_f16<<<grid, 128, 0, s>>>(A, sAb, lda, W, sWb, ldw, outF, outH,
                                       zdiv, sOo, sOi, ldo, bias, resid, M, N, K, act);
}

extern "C" void kernel_launch(void* const* d_in, const int* in_sizes, int n_in,
                              void* d_out, int out_size, void* d_ws, size_t ws_size,
                              hipStream_t stream) {
    (void)in_sizes; (void)n_in; (void)out_size; (void)ws_size;
    const float* src     = (const float*)d_in[0];
    const float* pre_w   = (const float*)d_in[1];
    const float* pre_b   = (const float*)d_in[2];
    const float* qkv_w   = (const float*)d_in[3];
    const float* qkv_b   = (const float*)d_in[4];
    const float* scale_p = (const float*)d_in[5];
    const float* riem_p  = (const float*)d_in[6];
    const float* grass_p = (const float*)d_in[7];
    const float* conv_w  = (const float*)d_in[8];
    const float* conv_b  = (const float*)d_in[9];
    const float* proj_w  = (const float*)d_in[10];
    const float* proj_b  = (const float*)d_in[11];
    const float* norm1_w = (const float*)d_in[12];
    const float* norm1_b = (const float*)d_in[13];
    const float* lin1_w  = (const float*)d_in[14];
    const float* lin1_b  = (const float*)d_in[15];
    const float* lin2_w  = (const float*)d_in[16];
    const float* lin2_b  = (const float*)d_in[17];

    float* out_src  = (float*)d_out;                               // [16,512,384]
    float* out_attn = (float*)d_out + (size_t)8192 * 384;          // [16,6,512,512]

    // ---- workspace carve-up (256B aligned), ~198 MB peak ----
    char* cur = (char*)d_ws;
    auto alloc = [&](size_t bytes) -> void* {
        void* p = (void*)cur;
        cur += (bytes + 255) & ~(size_t)255;
        return p;
    };
    _Float16* xh      = (_Float16*)alloc((size_t)8192 * 384 * 2);
    _Float16* wqkv_h  = (_Float16*)alloc((size_t)1152 * 384 * 2);
    _Float16* wproj_h = (_Float16*)alloc((size_t)384 * 384 * 2);
    _Float16* wlin1_h = (_Float16*)alloc((size_t)1152 * 384 * 2);
    _Float16* wlin2_h = (_Float16*)alloc((size_t)384 * 1152 * 2);
    _Float16* qkvo_h  = (_Float16*)alloc((size_t)8192 * 1152 * 2);
    _Float16* qh      = (_Float16*)alloc((size_t)96 * 512 * 64 * 2);
    _Float16* kh      = (_Float16*)alloc((size_t)96 * 512 * 64 * 2);
    _Float16* vTh     = (_Float16*)alloc((size_t)96 * 64 * 512 * 2);
    _Float16* qgrh    = (_Float16*)alloc((size_t)96 * 512 * 64 * 2);
    _Float16* kgrh    = (_Float16*)alloc((size_t)96 * 512 * 64 * 2);
    float*    qn2     = (float*)alloc((size_t)96 * 512 * 4);
    float*    kn2     = (float*)alloc((size_t)96 * 512 * 4);
    float*    Gq      = (float*)alloc((size_t)96 * 4096 * 4);
    float*    Gk      = (float*)alloc((size_t)96 * 4096 * 4);
    _Float16* RinvTq  = (_Float16*)alloc((size_t)96 * 4096 * 2);
    _Float16* RinvTk  = (_Float16*)alloc((size_t)96 * 4096 * 2);
    _Float16* qkh     = (_Float16*)alloc((size_t)96 * 512 * 512 * 2);  // reused as hbuf
    _Float16* grh     = (_Float16*)alloc((size_t)96 * 512 * 512 * 2);  // becomes attn f16
    _Float16* attno_h = (_Float16*)alloc((size_t)8192 * 384 * 2);
    float*    projb   = (float*)alloc((size_t)8192 * 384 * 4);
    float*    ubuf    = (float*)alloc((size_t)8192 * 384 * 4);
    _Float16* hbuf    = qkh;  // 18.9MB <= 50.3MB, qkh dead after mix_softmax

    // 1) weights -> f16
    cvt_f16_kernel<<<(1152 * 384 + 255) / 256, 256, 0, stream>>>(qkv_w, wqkv_h, 1152 * 384);
    cvt_f16_kernel<<<(384 * 384 + 255) / 256, 256, 0, stream>>>(proj_w, wproj_h, 384 * 384);
    cvt_f16_kernel<<<(1152 * 384 + 255) / 256, 256, 0, stream>>>(lin1_w, wlin1_h, 1152 * 384);
    cvt_f16_kernel<<<(384 * 1152 + 255) / 256, 256, 0, stream>>>(lin2_w, wlin2_h, 384 * 1152);

    // 2) pre-LN -> xh (f16)
    add_ln_kernel<<<8192, 128, 0, stream>>>(src, nullptr, pre_w, pre_b, xh, nullptr);

    // 3) QKV GEMM: [8192,384] @ [1152,384]^T + b -> qkvo_h f16   (TDM-staged)
    launch_gemm_tdm(stream, xh, 0, 384, wqkv_h, 0, 384, nullptr, qkvo_h,
                    1, 0, 0, 1152, qkv_b, nullptr, 8192, 1152, 384, 0, 1);

    // 4) split into q/k (n-major) and v (d-major) + norms
    qkv_rearrange_kernel<<<8192, 384, 0, stream>>>(qkvo_h, qh, kh, vTh, qn2, kn2);

    // 5) Cholesky-QR: Gram, factor+invert, then qgr = q @ R^{-1} (batched WMMA)
    gram64_kernel<<<96, 256, 0, stream>>>(qh, Gq);
    gram64_kernel<<<96, 256, 0, stream>>>(kh, Gk);
    chol_inv64_kernel<<<96, 64, 0, stream>>>(Gq, RinvTq);
    chol_inv64_kernel<<<96, 64, 0, stream>>>(Gk, RinvTk);
    launch_gemm(stream, qh, (long long)512 * 64, 64, RinvTq, (long long)64 * 64, 64,
                nullptr, qgrh, 1, (long long)512 * 64, 0, 64,
                nullptr, nullptr, 512, 64, 64, 0, 96);
    launch_gemm(stream, kh, (long long)512 * 64, 64, RinvTk, (long long)64 * 64, 64,
                nullptr, kgrh, 1, (long long)512 * 64, 0, 64,
                nullptr, nullptr, 512, 64, 64, 0, 96);

    // 6) score GEMMs (batched over 96 (b,h)): qk = q k^T, gr = qgr kgr^T (TDM-staged)
    launch_gemm_tdm(stream, qh, (long long)512 * 64, 64, kh, (long long)512 * 64, 64,
                    nullptr, qkh, 1, (long long)512 * 512, 0, 512,
                    nullptr, nullptr, 512, 512, 64, 0, 96);
    launch_gemm_tdm(stream, qgrh, (long long)512 * 64, 64, kgrh, (long long)512 * 64, 64,
                    nullptr, grh, 1, (long long)512 * 512, 0, 512,
                    nullptr, nullptr, 512, 512, 64, 0, 96);

    // 7) features + conv mix + softmax -> attn (f32 to d_out, f16 into grh)
    mix_softmax_kernel<<<dim3(512, 16), 256, 0, stream>>>(
        qkh, grh, qn2, kn2, conv_w, conv_b, scale_p, riem_p, grass_p, out_attn);

    // 8) attn @ V -> token-major f16 [8192,384] (split batch stride: b outer, h inner)
    launch_gemm(stream, grh, (long long)512 * 512, 512, vTh, (long long)64 * 512, 512,
                nullptr, attno_h, 6, (long long)512 * 384, 64, 384,
                nullptr, nullptr, 512, 64, 512, 0, 96);

    // 9) output projection -> projb f32   (TDM-staged)
    launch_gemm_tdm(stream, attno_h, 0, 384, wproj_h, 0, 384, projb, nullptr,
                    1, 0, 0, 384, proj_b, nullptr, 8192, 384, 384, 0, 1);

    // 10) residual + norm1 -> u (f32 in ubuf for final residual, f16 in xh for MLP)
    add_ln_kernel<<<8192, 128, 0, stream>>>(src, projb, norm1_w, norm1_b, xh, ubuf);

    // 11) MLP: lin1 + exact GELU -> hbuf f16; lin2 + bias + residual(u) -> d_out
    launch_gemm_tdm(stream, xh, 0, 384, wlin1_h, 0, 384, nullptr, hbuf,
                    1, 0, 0, 1152, lin1_b, nullptr, 8192, 1152, 384, 1, 1);
    launch_gemm_tdm(stream, hbuf, 0, 1152, wlin2_h, 0, 1152, out_src, nullptr,
                    1, 0, 0, 384, lin2_b, ubuf, 8192, 384, 1152, 0, 1);
}
